// HGNN_85770496901407
// MI455X (gfx1250) — matrix operand order repaired and checked by reading, not compiled, not run on hardware
//
#include <hip/hip_runtime.h>
#include <hip/hip_bf16.h>

// ---------------------------------------------------------------------------
// HGNN forward for MI455X (gfx1250, wave32, WMMA, async global->LDS).
// Transposed-feature formulation so every GEMM operand is K-contiguous:
//   t^T = de ⊙col (x^T @ A1)      (B tiles from A1^T rows)
//   u^T = t^T @ A1^T              (B tiles from A1 rows, bf16 out written
//                                  transposed -> vectorized b128 stores)
//   y^T = W^T @ u^T + b           (B tiles from u rows)
// A1 = dn ⊙ H in bf16 (82 MB) + its transpose; each L2-resident during use.
// Feature dim padded 400 -> 512 so ALL tiles are full (no guarded paths).
// ---------------------------------------------------------------------------

typedef __bf16 v16bf __attribute__((ext_vector_type(16)));
typedef __bf16 v8bf  __attribute__((ext_vector_type(8)));
typedef float  v8f   __attribute__((ext_vector_type(8)));
typedef int    v4i   __attribute__((ext_vector_type(4)));

#define N_NODES     6400
#define N_EDGES     6400
#define F_IN        400
#define DP0         512      // padded layer-0 feature dim (multiple of BM)
#define HDIM        256
#define BATCH       16
#define NODES_PER_B 400

#define BM 128
#define BN 64
#define BK 32
#define LDSP (BK + 8)        // padded LDS row (bf16) -> 80B pitch, 16B aligned

#if defined(__AMDGCN__) && __has_builtin(__builtin_amdgcn_global_load_async_to_lds_b128) && __has_builtin(__builtin_amdgcn_s_wait_asynccnt)
#define USE_ASYNC_LDS 1
#else
#define USE_ASYNC_LDS 0
#endif

__device__ __forceinline__ __bf16 f2bf(float f) {
  unsigned u = __builtin_bit_cast(unsigned, f);
  unsigned r = (u + 0x7FFFu + ((u >> 16) & 1u)) >> 16;   // round-to-nearest-even
  return __builtin_bit_cast(__bf16, (unsigned short)r);
}

// 16B global -> LDS copy: async path (ASYNCcnt) if available, else load+store.
__device__ __forceinline__ void cp16(const __bf16* g, __bf16* l) {
#if USE_ASYNC_LDS
  __builtin_amdgcn_global_load_async_to_lds_b128(
      (__attribute__((address_space(1))) v4i*)(const void*)g,
      (__attribute__((address_space(3))) v4i*)(void*)l, 0, 0);
#else
  *(uint4*)l = *(const uint4*)g;
#endif
}
__device__ __forceinline__ void async_fence() {
#if USE_ASYNC_LDS
  __builtin_amdgcn_s_wait_asynccnt(0);
#endif
}

// ------------------------------ degrees ------------------------------------
__global__ void node_deg_kernel(const float* __restrict__ H, float* __restrict__ dn, int E) {
  __shared__ float red[256];
  int row = blockIdx.x;
  const float* p = H + (size_t)row * E;
  float s = 0.f;
  for (int j = threadIdx.x; j < E; j += 256) s += p[j];
  red[threadIdx.x] = s;
  __syncthreads();
  for (int st = 128; st > 0; st >>= 1) {
    if (threadIdx.x < st) red[threadIdx.x] += red[threadIdx.x + st];
    __syncthreads();
  }
  if (threadIdx.x == 0) {
    float t = red[0];
    dn[row] = (t > 0.f) ? (1.0f / sqrtf(t)) : 0.f;
  }
}

__global__ void edge_deg_kernel(const float* __restrict__ H, float* __restrict__ de,
                                int Nn, int E) {
  int j = blockIdx.x * blockDim.x + threadIdx.x;
  if (j >= E) return;
  float s = 0.f;
  for (int i = 0; i < Nn; ++i) s += H[(size_t)i * E + j];
  de[j] = (s > 0.f) ? (1.f / s) : 0.f;
}

// ---------------- A1 = dn ⊙ H (bf16) and its transpose ---------------------
__global__ void scale_rows_to_bf16_kernel(const float* __restrict__ H,
                                          const float* __restrict__ dn,
                                          __bf16* __restrict__ out, int E) {
  int row = blockIdx.x;
  float s = dn[row];
  const float* src = H + (size_t)row * E;
  __bf16* dst = out + (size_t)row * E;
  for (int j = threadIdx.x; j < E; j += blockDim.x) dst[j] = f2bf(s * src[j]);
}

#define TT 64
__global__ void transpose_bf16_kernel(const __bf16* __restrict__ in,
                                      __bf16* __restrict__ out, int rows, int cols) {
  __shared__ __bf16 t[TT][TT + 4];
  int r0 = blockIdx.y * TT, c0 = blockIdx.x * TT;
  for (int idx = threadIdx.x; idx < TT * TT; idx += 256) {
    int r = idx >> 6, c = idx & 63;
    t[r][c] = in[(size_t)(r0 + r) * cols + c0 + c];
  }
  __syncthreads();
  for (int idx = threadIdx.x; idx < TT * TT; idx += 256) {
    int c = idx >> 6, r = idx & 63;
    out[(size_t)(c0 + c) * rows + r0 + r] = t[r][c];
  }
}

// ------------------------------ packing ------------------------------------
// xT[f][n] = f < F_IN ? bf16(x[n][f]) : 0        (grid = DP0 rows)
__global__ void pack_xT_kernel(const float* __restrict__ x, __bf16* __restrict__ xT,
                               int n, int f_real) {
  int f = blockIdx.x;
  const __bf16 BZ = __builtin_bit_cast(__bf16, (unsigned short)0);
  for (int j = threadIdx.x; j < n; j += blockDim.x)
    xT[(size_t)f * n + j] = (f < f_real) ? f2bf(x[(size_t)j * f_real + f]) : BZ;
}

// Wt[j][i] = i < kin_real ? bf16(W[i][j]) : 0    (grid = nout rows)
__global__ void pack_wT_kernel(const float* __restrict__ W, __bf16* __restrict__ Wt,
                               int kin_real, int kin_pad, int nout) {
  int j = blockIdx.x;
  const __bf16 BZ = __builtin_bit_cast(__bf16, (unsigned short)0);
  for (int i = threadIdx.x; i < kin_pad; i += blockDim.x)
    Wt[(size_t)j * kin_pad + i] = (i < kin_real) ? f2bf(W[(size_t)i * nout + j]) : BZ;
}

// ------------------------------ WMMA GEMM ----------------------------------
// C[M,N] = A[M,K](row-major, lda) @ B, where B is provided COLUMN-major as
// BT[N][K] row-major (ldb). All dims must be multiples of BM/BN/BK (padded).
// Epilogue: optional colScale[n], rowBias[m], relu; outputs:
//   Cf  row-major f32, Cb row-major bf16, CbT transposed bf16 [N][ldct] (b128).
__global__ __launch_bounds__(256)
void gemm_bf16(const __bf16* __restrict__ A, const __bf16* __restrict__ BT,
               float* __restrict__ Cf, __bf16* __restrict__ Cb, __bf16* __restrict__ CbT,
               const float* __restrict__ colScale, const float* __restrict__ rowBias,
               int M, int N, int K, int lda, int ldb, int ldc, int ldct, int doRelu) {
  __shared__ alignas(16) __bf16 As[2][BM][LDSP];
  __shared__ alignas(16) __bf16 Bs[2][BN][LDSP];

  const int tid  = threadIdx.x;
  const int lane = tid & 31;
  const int wave = tid >> 5;
  const int wr   = wave >> 1;            // 0..3 : 32-row strip
  const int wc   = wave & 1;             // 0..1 : 32-col strip
  const int m0   = blockIdx.y * BM;
  const int n0   = blockIdx.x * BN;

  v8f acc[2][2] = {};
  const int kTiles = K / BK;

  auto loadTile = [&](int buf, int k0) {
    // unconditional speculative L2 prefetch of the tile after this one
    // (OOB speculative prefetches are silently dropped; no branch in the loop)
    __builtin_prefetch((const void*)(A  + (size_t)(m0 + (tid >> 1)) * lda + k0 + BK + (tid & 1) * 16), 0, 1);
    __builtin_prefetch((const void*)(BT + (size_t)(n0 + (tid >> 2)) * ldb + k0 + BK + (tid & 3) * 8),  0, 1);
    // A tile: 128x32, k-contiguous: 512 chunks of 8 bf16 -> 2 per thread
    #pragma unroll
    for (int c = tid; c < (BM * BK) / 8; c += 256) {
      int m = c >> 2, k = (c & 3) << 3;
      cp16(A + (size_t)(m0 + m) * lda + k0 + k, &As[buf][m][k]);
    }
    // B tile: 64x32 from BT rows, k-contiguous: 256 chunks -> 1 per thread
    {
      int c = tid;
      int n = c >> 2, k = (c & 3) << 3;
      cp16(BT + (size_t)(n0 + n) * ldb + k0 + k, &Bs[buf][n][k]);
    }
  };

  auto compute = [&](int buf) {
    v16bf a[2], b[2];
    const int sub  = lane & 15;
    const int half = lane >> 4;
    // A fragment: lanes 0-15 K{0..7,16..23}, lanes 16-31 K{8..15,24..31}
    #pragma unroll
    for (int i = 0; i < 2; ++i) {
      const __bf16* p = &As[buf][wr * 32 + i * 16 + sub][half * 8];
      ((v8bf*)&a[i])[0] = *(const v8bf*)p;
      ((v8bf*)&a[i])[1] = *(const v8bf*)(p + 16);
    }
    // B fragment: lanes 0-15 K{0..7,16..23}, lanes 16-31 K{8..15,24..31} (col = lane&15)
    #pragma unroll
    for (int j = 0; j < 2; ++j) {
      const __bf16* p = &Bs[buf][wc * 32 + j * 16 + sub][half * 8];
      ((v8bf*)&b[j])[0] = *(const v8bf*)p;
      ((v8bf*)&b[j])[1] = *(const v8bf*)(p + 16);
    }
    #pragma unroll
    for (int i = 0; i < 2; ++i)
      #pragma unroll
      for (int j = 0; j < 2; ++j)
        acc[i][j] = __builtin_amdgcn_wmma_f32_16x16x32_bf16(
            false, a[i], false, b[j], (short)0, acc[i][j], false, false);
  };

  // pipeline: steady-state body is branch-free (load next, compute current)
  loadTile(0, 0);
  async_fence();
  __syncthreads();
  for (int kt = 0; kt < kTiles - 1; ++kt) {
    loadTile((kt + 1) & 1, (kt + 1) * BK);
    compute(kt & 1);
    async_fence();
    __syncthreads();
  }
  compute((kTiles - 1) & 1);

  // epilogue: C layout = 8 VGPRs, row = (lane>=16)*8 + r, col = lane&15
  const int sub  = lane & 15;
  const int half = lane >> 4;
  #pragma unroll
  for (int i = 0; i < 2; ++i) {
    #pragma unroll
    for (int j = 0; j < 2; ++j) {
      int n  = n0 + wc * 32 + j * 16 + sub;
      int mb = m0 + wr * 32 + i * 16 + half * 8;
      float cs = colScale ? colScale[n] : 1.f;
      float vals[8];
      #pragma unroll
      for (int r = 0; r < 8; ++r) {
        float v = acc[i][j][r] * cs;
        if (rowBias) v += rowBias[mb + r];
        if (doRelu)  v = fmaxf(v, 0.f);
        vals[r] = v;
      }
      if (Cf) {
        #pragma unroll
        for (int r = 0; r < 8; ++r) Cf[(size_t)(mb + r) * ldc + n] = vals[r];
      }
      if (Cb) {
        #pragma unroll
        for (int r = 0; r < 8; ++r) Cb[(size_t)(mb + r) * ldc + n] = f2bf(vals[r]);
      }
      if (CbT) {   // transposed bf16 out: 8 contiguous elements -> one b128
        unsigned short p[8];
        #pragma unroll
        for (int r = 0; r < 8; ++r)
          p[r] = __builtin_bit_cast(unsigned short, f2bf(vals[r]));
        *(uint4*)&CbT[(size_t)n * ldct + mb] = *(const uint4*)p;
      }
    }
  }
}

// ------------------------------ tail ---------------------------------------
// s[n] = sum_f yT[f][n]   (coalesced column sums)
__global__ void colsum_kernel(const float* __restrict__ yT, float* __restrict__ s,
                              int rows, int n) {
  int j = blockIdx.x * blockDim.x + threadIdx.x;
  if (j >= n) return;
  float v = 0.f;
  for (int i = 0; i < rows; ++i) v += yT[(size_t)i * n + j];
  s[j] = v;
}

__global__ void final_kernel(const float* __restrict__ s, const float* __restrict__ Wm,
                             const float* __restrict__ bm, float* __restrict__ out, int nn) {
  __shared__ float red[256];
  int b = blockIdx.x;
  float v = 0.f;
  for (int j = threadIdx.x; j < nn; j += 256) v += s[b * nn + j] * Wm[j];
  red[threadIdx.x] = v; __syncthreads();
  for (int st = 128; st > 0; st >>= 1) {
    if (threadIdx.x < st) red[threadIdx.x] += red[threadIdx.x + st];
    __syncthreads();
  }
  if (threadIdx.x == 0) out[b] = red[0] + bm[0];
}

// ------------------------------ driver -------------------------------------
extern "C" void kernel_launch(void* const* d_in, const int* in_sizes, int n_in,
                              void* d_out, int out_size, void* d_ws, size_t ws_size,
                              hipStream_t stream) {
  (void)in_sizes; (void)n_in; (void)out_size; (void)ws_size;
  const float* x  = (const float*)d_in[0];
  const float* H  = (const float*)d_in[1];
  const float* W0 = (const float*)d_in[2];
  const float* b0 = (const float*)d_in[3];
  const float* W1 = (const float*)d_in[4];
  const float* b1 = (const float*)d_in[5];
  const float* W2 = (const float*)d_in[6];
  const float* b2 = (const float*)d_in[7];
  const float* Wm = (const float*)d_in[10];
  const float* bm = (const float*)d_in[11];
  float* out = (float*)d_out;

  size_t off = 0;
  auto alloc = [&](size_t bytes) -> void* {
    void* p = (char*)d_ws + off;
    off += (bytes + 255) & ~(size_t)255;
    return p;
  };
  __bf16* A1b  = (__bf16*)alloc((size_t)N_NODES * N_EDGES * 2);  // [node][edge]
  __bf16* A1Tb = (__bf16*)alloc((size_t)N_EDGES * N_NODES * 2);  // [edge][node]
  float*  dn   = (float*)alloc((size_t)N_NODES * 4);
  float*  de   = (float*)alloc((size_t)N_EDGES * 4);
  __bf16* xT   = (__bf16*)alloc((size_t)DP0 * N_NODES * 2);      // [f][node]
  __bf16* tT   = (__bf16*)alloc((size_t)DP0 * N_EDGES * 2);      // [f][edge]
  __bf16* uN   = (__bf16*)alloc((size_t)N_NODES * DP0 * 2);      // [node][f]
  float*  yTf  = (float*)alloc((size_t)HDIM * N_NODES * 4);      // [j][node]
  __bf16* Wt0  = (__bf16*)alloc((size_t)HDIM * DP0 * 2);
  __bf16* Wt1  = (__bf16*)alloc((size_t)HDIM * HDIM * 2);
  __bf16* Wt2  = (__bf16*)alloc((size_t)HDIM * HDIM * 2);
  float*  sv   = (float*)alloc((size_t)N_NODES * 4);

  node_deg_kernel<<<N_NODES, 256, 0, stream>>>(H, dn, N_EDGES);
  edge_deg_kernel<<<(N_EDGES + 255) / 256, 256, 0, stream>>>(H, de, N_NODES, N_EDGES);
  scale_rows_to_bf16_kernel<<<N_NODES, 256, 0, stream>>>(H, dn, A1b, N_EDGES);
  transpose_bf16_kernel<<<dim3(N_NODES / TT, N_EDGES / TT), 256, 0, stream>>>(
      A1b, A1Tb, N_NODES, N_EDGES);
  pack_xT_kernel<<<DP0, 256, 0, stream>>>(x, xT, N_NODES, F_IN);
  pack_wT_kernel<<<HDIM, 256, 0, stream>>>(W0, Wt0, F_IN, DP0, HDIM);
  pack_wT_kernel<<<HDIM, 256, 0, stream>>>(W1, Wt1, HDIM, HDIM, HDIM);
  pack_wT_kernel<<<HDIM, 256, 0, stream>>>(W2, Wt2, HDIM, HDIM, HDIM);

  // layer order per reference: relu(W0), W2, relu(W1), W2
  const __bf16* Wseq[4] = { Wt0, Wt2, Wt1, Wt2 };
  const float*  bseq[4] = { b0,  b2,  b1,  b2  };
  const int     rseq[4] = { 1, 0, 1, 0 };

  int dP = DP0;
  for (int L = 0; L < 4; ++L) {
    // t^T = de ⊙col (x^T @ A1)      [dP, E]
    gemm_bf16<<<dim3(N_EDGES / BN, dP / BM), 256, 0, stream>>>(
        xT, A1Tb, nullptr, tT, nullptr, de, nullptr,
        dP, N_EDGES, N_NODES, N_NODES, N_NODES, N_EDGES, 0, 0);
    // u^T = t^T @ A1^T  -> written transposed as u [node][f] (b128 stores)
    gemm_bf16<<<dim3(N_NODES / BN, dP / BM), 256, 0, stream>>>(
        tT, A1b, nullptr, nullptr, uN, nullptr, nullptr,
        dP, N_NODES, N_EDGES, N_EDGES, N_EDGES, 0, DP0, 0);
    // y^T = relu?(W^T @ u^T + b)    [HDIM, node]  (f32 + bf16 row-major)
    gemm_bf16<<<dim3(N_NODES / BN, HDIM / BM), 256, 0, stream>>>(
        Wseq[L], uN, yTf, xT, nullptr, nullptr, bseq[L],
        HDIM, N_NODES, dP, dP, DP0, N_NODES, 0, rseq[L]);
    dP = HDIM;
  }

  // att == softmax([N,1]) == 1 -> per-node sum over features, then projection
  colsum_kernel<<<(N_NODES + 255) / 256, 256, 0, stream>>>(yTf, sv, HDIM, N_NODES);
  final_kernel<<<BATCH, 256, 0, stream>>>(sv, Wm, bm, out, NODES_PER_B);
}
